// MultiHeadAttention_85375359910073
// MI455X (gfx1250) — matrix operand order repaired and checked by reading, not compiled, not run on hardware
//
#include <hip/hip_runtime.h>

// ---------------- types / helpers ----------------
typedef __attribute__((ext_vector_type(16))) __bf16 v16bf;
typedef __attribute__((ext_vector_type(8)))  float  v8f;

struct alignas(16) U4 { unsigned int x, y, z, w; };
struct alignas(16) H8 { __bf16 h[8]; };
struct B32 { U4 a, b; };

static __device__ inline v16bf load_frag(const __bf16* c0, const __bf16* c1) {
  B32 t;
  t.a = *(const U4*)c0;
  t.b = *(const U4*)c1;
  return __builtin_bit_cast(v16bf, t);
}

static __device__ inline v8f wmma_bf16(v16bf a, v16bf b, v8f c) {
  // D(16x16 f32) = A(16x32 bf16) * B(32x16 bf16) + C
  return __builtin_amdgcn_wmma_f32_16x16x32_bf16(false, a, false, b, (short)0, c,
                                                 false, false);
}

#define LDA 40  // padded row length (elems) for 32-wide bf16 LDS tiles
#define NB   4
#define NT   2048
#define NC   1024
#define NH   16
#define ND   64

// ======================================================================
// Kernel 1: fused Q/K/V projections.
//   z=0: q = (Q@Wq+bq) * 0.125 -> qws  (B,H,T,64) bf16
//   z=1: k =  K@Wk+bk          -> kws  (B,H,T,64) bf16
//   z=2: v =  V@Wv+bv          -> vtws (B,H,64,T) bf16  (transposed)
// GEMM: M=8192, N=1024, K=1024.  WG = 256 thr = 8 waves, tile 128x128.
// Per wave: 32x64 (2x4 WMMA tiles) -> 8 WMMA per 32-K step.
// ======================================================================
__global__ __launch_bounds__(256) void qkv_proj_kernel(
    const float* __restrict__ Q, const float* __restrict__ K,
    const float* __restrict__ V,
    const float* __restrict__ Wq, const float* __restrict__ bq,
    const float* __restrict__ Wk, const float* __restrict__ bk,
    const float* __restrict__ Wv, const float* __restrict__ bv,
    __bf16* __restrict__ qws, __bf16* __restrict__ kws,
    __bf16* __restrict__ vtws) {
  const int z = blockIdx.z;
  const float* In   = (z == 0) ? Q  : (z == 1) ? K  : V;
  const float* W    = (z == 0) ? Wq : (z == 1) ? Wk : Wv;
  const float* bias = (z == 0) ? bq : (z == 1) ? bk : bv;

  __shared__ __bf16 As[128 * LDA];   // A tile: [m][k]
  __shared__ __bf16 Bt[128 * LDA];   // W^T tile: [n][k]

  const int tid  = threadIdx.x;
  const int lane = tid & 31, wave = tid >> 5;
  const int wm = wave & 3, wn = wave >> 2;     // 4x2 wave grid -> 128x128
  const int n16 = lane & 15, kg = lane >> 4;

  const int m0 = blockIdx.y * 128;
  const int n0 = blockIdx.x * 128;

  v8f acc[2][4];
  const v8f vzero = {0.f, 0.f, 0.f, 0.f, 0.f, 0.f, 0.f, 0.f};
#pragma unroll
  for (int i = 0; i < 2; i++)
#pragma unroll
    for (int j = 0; j < 4; j++) acc[i][j] = vzero;

  // staging geometry (per thread)
  const int ar = tid >> 1, ah = (tid & 1) << 4;       // A: row, 16-elem half
  const int wk = tid >> 3, wn0 = (tid & 7) << 4;      // W: k-row, 16 cols

  for (int kc = 0; kc < NC; kc += 32) {
    const float* asrc = In + (size_t)(m0 + ar) * NC + kc + ah;
    const float* wsrc = W + (size_t)(kc + wk) * NC + n0 + wn0;
    if (kc + 32 < NC) {  // prefetch next K tiles into L2/WGP$
      __builtin_prefetch(asrc + 32, 0, 1);
      __builtin_prefetch(wsrc + 32 * NC, 0, 1);
    }
    // ---- stage A tile: 128x32 fp32 -> bf16, 16 elems/thread ----
    {
      H8 pk;
#pragma unroll
      for (int i = 0; i < 8; i++) pk.h[i] = (__bf16)asrc[i];
      *(U4*)(As + ar * LDA + ah) = __builtin_bit_cast(U4, pk);
#pragma unroll
      for (int i = 0; i < 8; i++) pk.h[i] = (__bf16)asrc[8 + i];
      *(U4*)(As + ar * LDA + ah + 8) = __builtin_bit_cast(U4, pk);
    }
    // ---- stage W^T tile: 32x128 fp32 -> bf16 transposed, 16 elems/thread ----
    {
#pragma unroll
      for (int i = 0; i < 16; i++) Bt[(wn0 + i) * LDA + wk] = (__bf16)wsrc[i];
    }
    __syncthreads();

    v16bf af[2], bf_[4];
#pragma unroll
    for (int mi = 0; mi < 2; mi++) {
      const __bf16* p = As + (wm * 32 + mi * 16 + n16) * LDA + kg * 16;
      af[mi] = load_frag(p, p + 8);
    }
#pragma unroll
    for (int ni = 0; ni < 4; ni++) {
      const __bf16* p = Bt + (wn * 64 + ni * 16 + n16) * LDA + kg * 16;
      bf_[ni] = load_frag(p, p + 8);
    }
#pragma unroll
    for (int mi = 0; mi < 2; mi++)
#pragma unroll
      for (int ni = 0; ni < 4; ni++)
        acc[mi][ni] = wmma_bf16(af[mi], bf_[ni], acc[mi][ni]);
    __syncthreads();
  }

  // ---- epilogue: bias, (scale), convert, scatter into head layout ----
#pragma unroll
  for (int mi = 0; mi < 2; mi++)
#pragma unroll
    for (int ni = 0; ni < 4; ni++) {
      const int nc = n0 + wn * 64 + ni * 16 + n16;
      const float bcol = bias[nc];
      const int h = nc >> 6, d = nc & 63;
#pragma unroll
      for (int r = 0; r < 8; r++) {
        const int mr = m0 + wm * 32 + mi * 16 + r + 8 * kg;
        const int b = mr >> 11, t = mr & 2047;
        const float val = acc[mi][ni][r] + bcol;
        const int bh = b * NH + h;
        if (z == 0)
          qws[((size_t)bh * NT + t) * ND + d] = (__bf16)(val * 0.125f);
        else if (z == 1)
          kws[((size_t)bh * NT + t) * ND + d] = (__bf16)val;
        else
          vtws[((size_t)bh * ND + d) * NT + t] = (__bf16)val;
      }
    }
}

// ======================================================================
// Kernel 2: flash attention per (b,h).  Each wave: 16 query rows,
// streams keys in blocks of 32, online softmax, writes attn into the
// reference's buggy-reshape layout X (B, 2048, 1024) bf16.
// ======================================================================
__global__ __launch_bounds__(256) void attn_kernel(
    const __bf16* __restrict__ qws, const __bf16* __restrict__ kws,
    const __bf16* __restrict__ vtws, __bf16* __restrict__ X) {
  __shared__ __bf16 Pl[8][16 * LDA];  // per-wave P transpose scratch

  const int tid = threadIdx.x;
  const int lane = tid & 31, wave = tid >> 5;
  const int n16 = lane & 15, kg = lane >> 4;
  const int bh = blockIdx.y, b = bh >> 4, h = bh & 15;
  const int t0 = blockIdx.x * 128 + wave * 16;

  const __bf16* qbase = qws + (size_t)bh * NT * ND;
  const __bf16* kbase = kws + (size_t)bh * NT * ND;
  const __bf16* vbase = vtws + (size_t)bh * ND * NT;
  __bf16* Pw = &Pl[wave][0];

  // persistent Q fragments (d-steps 0..31 and 32..63)
  v16bf qf[2];
  {
    const __bf16* p = qbase + (size_t)(t0 + n16) * ND + kg * 16;
    qf[0] = load_frag(p, p + 8);
    qf[1] = load_frag(p + 32, p + 40);
  }

  const v8f vzero = {0.f, 0.f, 0.f, 0.f, 0.f, 0.f, 0.f, 0.f};
  float m[8], l[8];
  v8f O[4];
#pragma unroll
  for (int r = 0; r < 8; r++) { m[r] = -INFINITY; l[r] = 0.f; }
#pragma unroll
  for (int ni = 0; ni < 4; ni++) O[ni] = vzero;

  for (int j0 = 0; j0 < NT; j0 += 32) {
    // ---- scores: two 16x16 tiles over d=64 (2 WMMA each) ----
    v8f S[2];
#pragma unroll
    for (int jt = 0; jt < 2; jt++) {
      const __bf16* kp = kbase + (size_t)(j0 + jt * 16 + n16) * ND;
      v16bf kf0 = load_frag(kp + kg * 16, kp + kg * 16 + 8);
      v16bf kf1 = load_frag(kp + 32 + kg * 16, kp + 32 + kg * 16 + 8);
      S[jt] = wmma_bf16(qf[0], kf0, vzero);
      S[jt] = wmma_bf16(qf[1], kf1, S[jt]);
    }

    // ---- online softmax over the 32 new columns ----
    float bm[8], rs[8];
#pragma unroll
    for (int r = 0; r < 8; r++) bm[r] = fmaxf(S[0][r], S[1][r]);
#pragma unroll
    for (int off = 1; off < 16; off <<= 1)
#pragma unroll
      for (int r = 0; r < 8; r++)
        bm[r] = fmaxf(bm[r], __shfl_xor(bm[r], off, 32));

#pragma unroll
    for (int r = 0; r < 8; r++) {
      const float mn = fmaxf(m[r], bm[r]);
      const float alpha = __expf(m[r] - mn);
      m[r] = mn;
      const float p0 = __expf(S[0][r] - mn);
      const float p1 = __expf(S[1][r] - mn);
      rs[r] = p0 + p1;
      const int row = r + 8 * kg;
      Pw[row * LDA + n16]      = (__bf16)p0;
      Pw[row * LDA + 16 + n16] = (__bf16)p1;
      l[r] *= alpha;
#pragma unroll
      for (int ni = 0; ni < 4; ni++) O[ni][r] *= alpha;
    }
#pragma unroll
    for (int off = 1; off < 16; off <<= 1)
#pragma unroll
      for (int r = 0; r < 8; r++) rs[r] += __shfl_xor(rs[r], off, 32);
#pragma unroll
    for (int r = 0; r < 8; r++) l[r] += rs[r];

    // C-layout -> A-layout transpose of P through wave-private LDS
    asm volatile("s_wait_dscnt 0x0" ::: "memory");
    const __bf16* pp = Pw + n16 * LDA + kg * 16;
    v16bf pf = load_frag(pp, pp + 8);

    // ---- O += P @ V  (4 N-tiles over d=64) ----
#pragma unroll
    for (int ni = 0; ni < 4; ni++) {
      const __bf16* vp = vbase + (size_t)(ni * 16 + n16) * NT + j0 + kg * 16;
      v16bf vf = load_frag(vp, vp + 8);
      O[ni] = wmma_bf16(pf, vf, O[ni]);
    }
  }

  // ---- normalize and scatter into buggy-reshape X ----
  float rl[8];
#pragma unroll
  for (int r = 0; r < 8; r++) rl[r] = 1.0f / l[r];
#pragma unroll
  for (int ni = 0; ni < 4; ni++)
#pragma unroll
    for (int r = 0; r < 8; r++) {
      const int t = t0 + r + 8 * kg;
      const int d = ni * 16 + n16;
      const int tp = h * 128 + (t >> 4);
      const int cp = ((t & 15) << 6) + d;
      X[((size_t)(b * NT + tp)) * NC + cp] = (__bf16)(O[ni][r] * rl[r]);
    }
}

// ======================================================================
// Kernel 3: out = X(8192x1024 bf16) @ Wo(1024x1024 f32->bf16) + bo, fp32 out.
// Same 128x128 WG tile / 32x64 wave tile as kernel 1.
// ======================================================================
__global__ __launch_bounds__(256) void out_proj_kernel(
    const __bf16* __restrict__ X, const float* __restrict__ Wo,
    const float* __restrict__ bo, float* __restrict__ out) {
  __shared__ __bf16 As[128 * LDA];
  __shared__ __bf16 Bt[128 * LDA];

  const int tid = threadIdx.x;
  const int lane = tid & 31, wave = tid >> 5;
  const int wm = wave & 3, wn = wave >> 2;
  const int n16 = lane & 15, kg = lane >> 4;

  const int m0 = blockIdx.y * 128;
  const int n0 = blockIdx.x * 128;

  v8f acc[2][4];
  const v8f vzero = {0.f, 0.f, 0.f, 0.f, 0.f, 0.f, 0.f, 0.f};
#pragma unroll
  for (int i = 0; i < 2; i++)
#pragma unroll
    for (int j = 0; j < 4; j++) acc[i][j] = vzero;

  const int ar = tid >> 1, ah = (tid & 1) << 4;
  const int wk = tid >> 3, wn0 = (tid & 7) << 4;

  for (int kc = 0; kc < NC; kc += 32) {
    const __bf16* asrc = X + (size_t)(m0 + ar) * NC + kc + ah;
    const float* wsrc = Wo + (size_t)(kc + wk) * NC + n0 + wn0;
    if (kc + 32 < NC) {
      __builtin_prefetch(asrc + 32, 0, 1);
      __builtin_prefetch(wsrc + 32 * NC, 0, 1);
    }
    {
      *(U4*)(As + ar * LDA + ah)     = *(const U4*)asrc;
      *(U4*)(As + ar * LDA + ah + 8) = *(const U4*)(asrc + 8);
    }
    {
#pragma unroll
      for (int i = 0; i < 16; i++) Bt[(wn0 + i) * LDA + wk] = (__bf16)wsrc[i];
    }
    __syncthreads();

    v16bf af[2], bf_[4];
#pragma unroll
    for (int mi = 0; mi < 2; mi++) {
      const __bf16* p = As + (wm * 32 + mi * 16 + n16) * LDA + kg * 16;
      af[mi] = load_frag(p, p + 8);
    }
#pragma unroll
    for (int ni = 0; ni < 4; ni++) {
      const __bf16* p = Bt + (wn * 64 + ni * 16 + n16) * LDA + kg * 16;
      bf_[ni] = load_frag(p, p + 8);
    }
#pragma unroll
    for (int mi = 0; mi < 2; mi++)
#pragma unroll
      for (int ni = 0; ni < 4; ni++)
        acc[mi][ni] = wmma_bf16(af[mi], bf_[ni], acc[mi][ni]);
    __syncthreads();
  }

#pragma unroll
  for (int mi = 0; mi < 2; mi++)
#pragma unroll
    for (int ni = 0; ni < 4; ni++) {
      const int nc = n0 + wn * 64 + ni * 16 + n16;
      const float bcol = bo[nc];
#pragma unroll
      for (int r = 0; r < 8; r++) {
        const int mr = m0 + wm * 32 + mi * 16 + r + 8 * kg;
        out[(size_t)mr * NC + nc] = acc[mi][ni][r] + bcol;
      }
    }
}

// ======================================================================
extern "C" void kernel_launch(void* const* d_in, const int* in_sizes, int n_in,
                              void* d_out, int out_size, void* d_ws,
                              size_t ws_size, hipStream_t stream) {
  (void)in_sizes; (void)n_in; (void)out_size; (void)ws_size;
  const float* Q  = (const float*)d_in[0];
  const float* K  = (const float*)d_in[1];
  const float* V  = (const float*)d_in[2];
  const float* Wq = (const float*)d_in[3];
  const float* bq = (const float*)d_in[4];
  const float* Wk = (const float*)d_in[5];
  const float* bk = (const float*)d_in[6];
  const float* Wv = (const float*)d_in[7];
  const float* bv = (const float*)d_in[8];
  const float* Wo = (const float*)d_in[9];
  const float* bo = (const float*)d_in[10];

  char* ws = (char*)d_ws;
  const size_t SEG = (size_t)16 * 1024 * 1024;  // 16 MB per intermediate
  __bf16* qws  = (__bf16*)(ws + 0 * SEG);       // (B,H,T,64)
  __bf16* kws  = (__bf16*)(ws + 1 * SEG);       // (B,H,T,64)
  __bf16* vtws = (__bf16*)(ws + 2 * SEG);       // (B,H,64,T)
  __bf16* X    = (__bf16*)(ws + 3 * SEG);       // (B,2048,1024) buggy reshape

  dim3 blk(256, 1, 1);
  qkv_proj_kernel<<<dim3(8, 64, 3), blk, 0, stream>>>(
      Q, K, V, Wq, bq, Wk, bk, Wv, bv, qws, kws, vtws);
  attn_kernel<<<dim3(16, 64, 1), blk, 0, stream>>>(qws, kws, vtws, X);
  out_proj_kernel<<<dim3(8, 64, 1), blk, 0, stream>>>(X, Wo, bo,
                                                      (float*)d_out);
}
